// RelGATModel_34522947125453
// MI455X (gfx1250) — matrix operand order repaired and checked by reading, not compiled, not run on hardware
//
#include <hip/hip_runtime.h>
#include <hip/hip_bf16.h>
#include <math.h>

// ---------------- problem constants (match reference) ----------------
#define NNODES   50000
#define INDIM    1024
#define HEADS    4
#define OUTD     200
#define HID      800      // HEADS*OUTD
#define NREL     40
#define NEG_SLOPE 0.2f

// GEMM tiling: per-wave macro tile = 32 rows x 80 cols (2 M-tiles x 5 N-tiles)
#define MT 2
#define NT 5
#define ROWS_PW (MT * 16)            // 32
#define COLS_PW (NT * 16)            // 80
#define MBLOCKS ((NNODES + ROWS_PW - 1) / ROWS_PW)   // 1563
#define NPAD    (MBLOCKS * ROWS_PW)                   // 50016 (A padded rows)
#define NBLOCKS (HID / COLS_PW)                       // 10

typedef __attribute__((ext_vector_type(8)))  __bf16 v8bf;
typedef __attribute__((ext_vector_type(16))) __bf16 v16bf;
typedef __attribute__((ext_vector_type(8)))  float  v8f;

// round-to-nearest-even f32 -> bf16 (bit manipulation; no scalar __bf16 math)
__device__ __forceinline__ unsigned short f2bf(float f) {
    unsigned u = __float_as_uint(f);
    unsigned r = u + 0x7FFFu + ((u >> 16) & 1u);
    return (unsigned short)(r >> 16);
}

// order-preserving float -> uint key for atomicMax-based segment max
__device__ __forceinline__ unsigned fkey(float f) {
    unsigned u = __float_as_uint(f);
    return (u & 0x80000000u) ? ~u : (u | 0x80000000u);
}
__device__ __forceinline__ float funkey(unsigned k) {
    unsigned u = (k & 0x80000000u) ? (k & 0x7FFFFFFFu) : ~k;
    return __uint_as_float(u);
}

// ---- WMMA fragment loaders (ISA 16-bit operand layouts) ----
// A frag: elems 0-7 -> K = khalf*8 + i ; elems 8-15 -> K = 16 + khalf*8 + i
__device__ __forceinline__ v16bf load_afrag(const unsigned short* p, int kb, int khalf) {
    v8bf a0 = *(const v8bf*)(p + kb + khalf * 8);
    v8bf a1 = *(const v8bf*)(p + kb + 16 + khalf * 8);
    return __builtin_shufflevector(a0, a1, 0,1,2,3,4,5,6,7,8,9,10,11,12,13,14,15);
}
// B frag (from W^T row = output column): elems 0-7 -> K = khalf*8 + i ; elems 8-15 -> K = 16 + khalf*8 + i
__device__ __forceinline__ v16bf load_bfrag(const unsigned short* p, int kb, int khalf) {
    v8bf b0 = *(const v8bf*)(p + kb + khalf * 8);
    v8bf b1 = *(const v8bf*)(p + kb + 16 + khalf * 8);
    return __builtin_shufflevector(b0, b1, 0,1,2,3,4,5,6,7,8,9,10,11,12,13,14,15);
}

// ---------------- init / convert ----------------
__global__ void zero_f32_kernel(float* __restrict__ p, long long n) {
    long long i = (long long)blockIdx.x * blockDim.x + threadIdx.x;
    if (i < n) p[i] = 0.0f;
}

// converts node_emb -> bf16 and zero-fills the 16 pad rows (keeps NaN out of WMMA)
__global__ void cvt_node_kernel(const float* __restrict__ src,
                                unsigned short* __restrict__ dst) {
    long long i = (long long)blockIdx.x * blockDim.x + threadIdx.x;
    if (i < (long long)NPAD * INDIM)
        dst[i] = (i < (long long)NNODES * INDIM) ? f2bf(src[i]) : (unsigned short)0;
}

// Wt[c][k] = W[k][c]  (transpose so B-tiles load like A-tiles)
__global__ void cvt_wt_kernel(const float* __restrict__ W,
                              unsigned short* __restrict__ wt) {
    int i = blockIdx.x * blockDim.x + threadIdx.x;
    if (i < HID * INDIM) {
        int c = i / INDIM, k = i % INDIM;
        wt[i] = f2bf(W[(long long)k * HID + c]);
    }
}

// ---------------- WMMA GEMM: h[N,HID] = node_bf16[N,K] * W[K,HID] ----------------
// One wave per 32x80 macro tile: 10 accumulators, 10 WMMA per k-step against
// 7 fragment loads (2 A + 5 B) -> 320 v_wmma per wave, ~3.5 GB total L2 traffic.
__global__ __launch_bounds__(32) void gemm_bf16_wmma_kernel(
        const unsigned short* __restrict__ A,   // [NPAD][INDIM]   bf16 (padded)
        const unsigned short* __restrict__ Bt,  // [HID][INDIM]    bf16 (W^T)
        float* __restrict__ C) {                // [NNODES][HID]   f32
    const int lane  = threadIdx.x;        // 0..31, wave32
    const int khalf = lane >> 4;          // 0 or 1
    const int lrow  = lane & 15;

    const unsigned short* arow0 = A + (long long)(blockIdx.x * ROWS_PW + lrow) * INDIM;
    const unsigned short* arow1 = arow0 + (long long)16 * INDIM;
    const int colbase = blockIdx.y * COLS_PW;
    const unsigned short* brow[NT];
    #pragma unroll
    for (int t = 0; t < NT; ++t)
        brow[t] = Bt + (long long)(colbase + t * 16 + lrow) * INDIM;

    v8f acc[MT][NT];
    #pragma unroll
    for (int m = 0; m < MT; ++m)
        #pragma unroll
        for (int t = 0; t < NT; ++t) acc[m][t] = (v8f){};

    for (int kb = 0; kb < INDIM; kb += 32) {
        v16bf a0 = load_afrag(arow0, kb, khalf);
        v16bf a1 = load_afrag(arow1, kb, khalf);
        __builtin_prefetch(arow0 + kb + 64, 0, 0);   // global_prefetch_b8 next k-slab
        v16bf bfr[NT];
        #pragma unroll
        for (int t = 0; t < NT; ++t) bfr[t] = load_bfrag(brow[t], kb, khalf);
        #pragma unroll
        for (int t = 0; t < NT; ++t) {
            acc[0][t] = __builtin_amdgcn_wmma_f32_16x16x32_bf16(false, a0, false, bfr[t],
                                                                (short)0, acc[0][t], false, false);
            acc[1][t] = __builtin_amdgcn_wmma_f32_16x16x32_bf16(false, a1, false, bfr[t],
                                                                (short)0, acc[1][t], false, false);
        }
    }
    // C layout: VGPR j -> M = khalf*8 + j, N = lane&15. Guard pad rows (post-WMMA).
    #pragma unroll
    for (int m = 0; m < MT; ++m) {
        const int rowbase = blockIdx.x * ROWS_PW + m * 16 + khalf * 8;
        #pragma unroll
        for (int j = 0; j < 8; ++j) {
            const int r = rowbase + j;
            if (r < NNODES) {
                #pragma unroll
                for (int t = 0; t < NT; ++t)
                    C[(long long)r * HID + colbase + t * 16 + lrow] = acc[m][t][j];
            }
        }
    }
}

// ---------------- per-node attention scalars: s_src[n,h], s_dst[n,h] ----------------
__global__ void node_logit_kernel(const float* __restrict__ h,
                                  const float* __restrict__ a_src,
                                  const float* __restrict__ a_dst,
                                  float* __restrict__ s_src,
                                  float* __restrict__ s_dst) {
    long long idx = (long long)blockIdx.x * blockDim.x + threadIdx.x;
    if (idx >= (long long)NNODES * HEADS) return;
    int hh = (int)(idx & 3);
    long long n = idx >> 2;
    const float* hp = h + n * HID + hh * OUTD;
    const float* as = a_src + hh * OUTD;
    const float* ad = a_dst + hh * OUTD;
    float ss = 0.f, sd = 0.f;
    for (int j = 0; j < OUTD; ++j) { float v = hp[j]; ss += v * as[j]; sd += v * ad[j]; }
    s_src[idx] = ss; s_dst[idx] = sd;
}

// rel_logit[r,h] = <rel_feat[r, h*200: ], a_rel[h]>
__global__ void rel_logit_kernel(const float* __restrict__ rel_feat,
                                 const float* __restrict__ a_rel,
                                 float* __restrict__ rlog) {
    int idx = blockIdx.x * blockDim.x + threadIdx.x;
    if (idx >= NREL * HEADS) return;
    int hh = idx & 3, r = idx >> 2;
    const float* rp = rel_feat + (long long)r * HID + hh * OUTD;
    const float* ar = a_rel + hh * OUTD;
    float s = 0.f;
    for (int j = 0; j < OUTD; ++j) s += rp[j] * ar[j];
    rlog[idx] = s;
}

// ---------------- edge phase ----------------
__global__ void edge_logit_kernel(const int* __restrict__ esrc, const int* __restrict__ edst,
                                  const int* __restrict__ etype,
                                  const float* __restrict__ s_src, const float* __restrict__ s_dst,
                                  const float* __restrict__ rlog,
                                  float* __restrict__ elog, unsigned* __restrict__ segmax, int E) {
    int e = blockIdx.x * blockDim.x + threadIdx.x;
    if (e >= E) return;
    int s = esrc[e], d = edst[e], t = etype[e];
    #pragma unroll
    for (int hh = 0; hh < HEADS; ++hh) {
        float l = s_src[s * 4 + hh] + s_dst[d * 4 + hh] + rlog[t * 4 + hh];
        l = (l > 0.f) ? l : NEG_SLOPE * l;            // leaky relu
        elog[(long long)e * 4 + hh] = l;
        atomicMax(&segmax[d * 4 + hh], fkey(l));
    }
}

__global__ void edge_exp_kernel(const int* __restrict__ edst,
                                float* __restrict__ elog,
                                const unsigned* __restrict__ segmax,
                                float* __restrict__ denom, int E) {
    int e = blockIdx.x * blockDim.x + threadIdx.x;
    if (e >= E) return;
    int d = edst[e];
    #pragma unroll
    for (int hh = 0; hh < HEADS; ++hh) {
        float m  = funkey(segmax[d * 4 + hh]);
        float ev = __expf(elog[(long long)e * 4 + hh] - m);
        elog[(long long)e * 4 + hh] = ev;
        atomicAdd(&denom[d * 4 + hh], ev);
    }
}

// out[dst] += alpha * h[src]; one 256-thread block per edge, atomics land in L2
__global__ void scatter_kernel(const int* __restrict__ esrc, const int* __restrict__ edst,
                               const float* __restrict__ elog, const float* __restrict__ denom,
                               const float* __restrict__ h, float* __restrict__ out_acc, int E) {
    int e = blockIdx.x;
    if (e >= E) return;
    int s = esrc[e], d = edst[e];
    __shared__ float alpha[HEADS];
    if (threadIdx.x < HEADS)
        alpha[threadIdx.x] = elog[(long long)e * 4 + threadIdx.x] / denom[d * 4 + threadIdx.x];
    __syncthreads();
    const float* hs = h + (long long)s * HID;
    float* op = out_acc + (long long)d * HID;
    for (int idx = threadIdx.x; idx < HID; idx += blockDim.x)
        atomicAdd(&op[idx], alpha[idx / OUTD] * hs[idx]);
}

// ---------------- DistMult scoring: one wave (32 lanes) per triple ----------------
__global__ void score_kernel(const float* __restrict__ out_acc, const float* __restrict__ bias,
                             const float* __restrict__ rel_emb,
                             const int* __restrict__ sid, const int* __restrict__ rid,
                             const int* __restrict__ did,
                             float* __restrict__ scores, int B) {
    int wave = threadIdx.x >> 5, lane = threadIdx.x & 31;
    int b = blockIdx.x * (blockDim.x >> 5) + wave;
    if (b >= B) return;                                  // wave-uniform
    const float* sv = out_acc + (long long)sid[b] * HID;
    const float* dv = out_acc + (long long)did[b] * HID;
    const float* rv = rel_emb + (long long)rid[b] * HID;
    float acc = 0.f;
    for (int k = lane; k < HID; k += 32)
        acc += (sv[k] + bias[k]) * rv[k] * (dv[k] + bias[k]);
    #pragma unroll
    for (int off = 16; off > 0; off >>= 1) acc += __shfl_xor(acc, off, 32);
    if (lane == 0) scores[b] = acc;
}

// ---------------- launch ----------------
extern "C" void kernel_launch(void* const* d_in, const int* in_sizes, int n_in,
                              void* d_out, int out_size, void* d_ws, size_t ws_size,
                              hipStream_t stream) {
    const float* node_emb = (const float*)d_in[0];
    const float* W        = (const float*)d_in[1];
    const float* bias     = (const float*)d_in[2];
    const float* a_src    = (const float*)d_in[3];
    const float* a_dst    = (const float*)d_in[4];
    const float* a_rel    = (const float*)d_in[5];
    const float* rel_feat = (const float*)d_in[6];
    const float* rel_emb  = (const float*)d_in[7];
    const int*   eidx     = (const int*)d_in[8];
    const int*   etype    = (const int*)d_in[9];
    const int*   src_ids  = (const int*)d_in[10];
    const int*   rel_ids  = (const int*)d_in[11];
    const int*   dst_ids  = (const int*)d_in[12];

    const int E  = in_sizes[9];
    const int Bq = in_sizes[10];
    const int* esrc = eidx;
    const int* edst = eidx + E;

    // ---- workspace carve-up (all sizes 256B-multiples) ----
    char* ws = (char*)d_ws;
    size_t off = 0;
    auto carve = [&](size_t bytes) { void* p = ws + off; off += (bytes + 255) & ~(size_t)255; return p; };
    float*          h        = (float*)carve((size_t)NNODES * HID * 4);        // 160 MB
    float*          out_acc  = (float*)carve((size_t)NNODES * HID * 4);        // 160 MB
    unsigned*       segmax   = (unsigned*)carve((size_t)NNODES * HEADS * 4);   // 0.8 MB
    float*          denom    = (float*)carve((size_t)NNODES * HEADS * 4);      // 0.8 MB
    float*          s_src    = (float*)carve((size_t)NNODES * HEADS * 4);
    float*          s_dst    = (float*)carve((size_t)NNODES * HEADS * 4);
    float*          rlog     = (float*)carve((size_t)NREL * HEADS * 4);
    float*          elog     = (float*)carve((size_t)E * HEADS * 4);           // 1.6 MB
    unsigned short* node_bf  = (unsigned short*)carve((size_t)NPAD * INDIM * 2);   // ~102 MB
    unsigned short* wt_bf    = (unsigned short*)carve((size_t)HID * INDIM * 2);    // 1.6 MB

    // 1) zero out_acc + segmax + denom (contiguous region)
    {
        long long n = (long long)NNODES * HID + 2LL * NNODES * HEADS;
        zero_f32_kernel<<<(unsigned)((n + 255) / 256), 256, 0, stream>>>(out_acc, n);
    }
    // 2) bf16 conversions (node conversion also zero-fills pad rows)
    {
        long long n = (long long)NPAD * INDIM;
        cvt_node_kernel<<<(unsigned)((n + 255) / 256), 256, 0, stream>>>(node_emb, node_bf);
        cvt_wt_kernel<<<(HID * INDIM + 255) / 256, 256, 0, stream>>>(W, wt_bf);
    }
    // 3) WMMA GEMM: 1563 x 10 waves, 32x80 macro tile each (320 v_wmma/wave)
    gemm_bf16_wmma_kernel<<<dim3(MBLOCKS, NBLOCKS), 32, 0, stream>>>(node_bf, wt_bf, h);
    // 4) per-node / per-rel attention scalars
    {
        long long n = (long long)NNODES * HEADS;
        node_logit_kernel<<<(unsigned)((n + 255) / 256), 256, 0, stream>>>(h, a_src, a_dst, s_src, s_dst);
        rel_logit_kernel<<<1, 256, 0, stream>>>(rel_feat, a_rel, rlog);
    }
    // 5) edge logits + segment max
    edge_logit_kernel<<<(E + 255) / 256, 256, 0, stream>>>(esrc, edst, etype, s_src, s_dst, rlog, elog, segmax, E);
    // 6) exp + denominators
    edge_exp_kernel<<<(E + 255) / 256, 256, 0, stream>>>(edst, elog, segmax, denom, E);
    // 7) weighted scatter-aggregate (atomics resolve in 192MB L2; out_acc is resident)
    scatter_kernel<<<E, 256, 0, stream>>>(esrc, edst, elog, denom, h, out_acc, E);
    // 8) DistMult scores, wave-per-triple
    score_kernel<<<(Bq + 7) / 8, 256, 0, stream>>>(out_acc, bias, rel_emb, src_ids, rel_ids, dst_ids,
                                                   (float*)d_out, Bq);
}
